// Model_15487652069579
// MI455X (gfx1250) — compile-verified, matrix operand order checked by reading, not run on hardware
//
#include <hip/hip_runtime.h>

// MPNN (NNConv + GRU) for MI455X / gfx1250, wave32.
// Strategy: never materialize the 655MB per-edge weight tensor. Fuse
//   m[e,o] = sum_{i,k} h[e,i]*t[e,k]*W2[(i*64+o),k] + sum_i h[e,i]*b2[i*64+o]
// as a bf16 WMMA GEMM per 16-edge tile with K = 64*128 + 64 = 8256,
// B = 1MB repacked en_W2 (L2-resident, step-invariant).

#define V_N     20000
#define E_N     40000
#define NODE_IN 74
#define EDGE_IN 12
#define D       64
#define EH      128
#define STEPS   6
#define KTOT    (D*EH + D)      // 8256 (bilinear + bias extension)
#define KTILES  (KTOT/32)       // 258
#define NTILES  (D/16)          // 4

typedef __attribute__((ext_vector_type(16))) __bf16 v16bf;
typedef __attribute__((ext_vector_type(8)))  float  v8f;

// ---------------- node projection: h = relu(node @ projW^T + b) ----------------
__global__ void __launch_bounds__(256)
proj_kernel(const float* __restrict__ node, const float* __restrict__ W,
            const float* __restrict__ b, float* __restrict__ h)
{
    const int tid = blockIdx.x * 256 + threadIdx.x;   // V_N*D threads exactly
    const int v = tid >> 6, j = tid & 63;
    float s = b[j];
    #pragma unroll 2
    for (int i = 0; i < NODE_IN; ++i)
        s += node[(size_t)v * NODE_IN + i] * W[j * NODE_IN + i];
    h[tid] = s > 0.f ? s : 0.f;
}

// ---------------- edge hidden: t = relu(edge @ W1^T + b1) ----------------
__global__ void __launch_bounds__(256)
edge_hidden_kernel(const float* __restrict__ edge, const float* __restrict__ W1,
                   const float* __restrict__ b1, float* __restrict__ t)
{
    const int tid = blockIdx.x * 256 + threadIdx.x;   // E_N*EH threads exactly
    const int e = tid >> 7, j = tid & 127;
    float s = b1[j];
    #pragma unroll
    for (int i = 0; i < EDGE_IN; ++i)
        s += edge[(size_t)e * EDGE_IN + i] * W1[j * EDGE_IN + i];
    t[tid] = s > 0.f ? s : 0.f;
}

// ------------- pack en_W2 (+ bias extension) into WMMA B-fragment order -------------
// Tile (kt, nt) holds B[kt*32 .. +31, nt*16 .. +15] as 512 bf16:
//   element q = lane*16 + j;  n = lane&15; khalf = lane>>4;
//   k_local = j + (j&8) + khalf*8  (16-bit operand fragment layout, ISA 7.12.2)
__global__ void __launch_bounds__(256)
pack_b_kernel(const float* __restrict__ W2, const float* __restrict__ b2,
              __bf16* __restrict__ Bp)
{
    const int tid  = blockIdx.x * 256 + threadIdx.x;  // KTILES*NTILES*512 exactly
    const int q    = tid & 511;
    const int tile = tid >> 9;
    const int nt   = tile & (NTILES - 1);
    const int kt   = tile / NTILES;
    const int lane = q >> 4, j = q & 15;
    const int n = lane & 15, khalf = lane >> 4;
    const int c = kt * 32 + j + (j & 8) + khalf * 8;  // composite K index
    const int o = nt * 16 + n;
    float val;
    if (c < D * EH) {
        const int i = c >> 7, k = c & 127;            // c = i*128 + k
        val = W2[(size_t)(i * D + o) * EH + k];
    } else {
        val = b2[(c - D * EH) * D + o];               // bias extension rows
    }
    Bp[tid] = (__bf16)val;
}

__global__ void __launch_bounds__(256)
zero_kernel(float* __restrict__ p, int n)
{
    const int tid = blockIdx.x * 256 + threadIdx.x;
    if (tid < n) p[tid] = 0.f;
}

// ------------- fused NNConv message + scatter: the WMMA kernel -------------
// Block = 128 threads (4 waves), 64 edges. Each wave: 16 edges x 64 outputs,
// 4 f32 accumulators, K looped in 258 chunks of 32 (bf16 16x16x32 WMMA).
__global__ void __launch_bounds__(128)
edge_msg_kernel(const float* __restrict__ h, const float* __restrict__ t,
                const __bf16* __restrict__ Bpack,
                const int* __restrict__ src, const int* __restrict__ dst,
                float* __restrict__ neigh)
{
    __shared__ float h_s[64 * D];    // gathered h[src] rows, 16 KB
    __shared__ float t_s[64 * EH];   // edge-hidden rows,    32 KB
    const int e0 = blockIdx.x * 64;

    for (int idx = threadIdx.x; idx < 64 * D; idx += 128) {
        const int el = idx >> 6, f = idx & 63;
        h_s[idx] = h[(size_t)src[e0 + el] * D + f];
    }
    for (int idx = threadIdx.x; idx < 64 * EH; idx += 128) {
        const int el = idx >> 7, f = idx & 127;
        t_s[idx] = t[(size_t)(e0 + el) * EH + f];
    }
    __syncthreads();

    const int wid   = threadIdx.x >> 5;
    const int lane  = threadIdx.x & 31;
    const int row   = lane & 15;
    const int khalf = lane >> 4;
    const int el    = wid * 16 + row;          // A-matrix row = edge in block

    v8f acc[NTILES] = {};
    const v16bf* __restrict__ Bfrag = (const v16bf*)Bpack;  // 32 frags/tile

    for (int kt = 0; kt < KTILES; ++kt) {
        // Build A fragment on the fly: A[e, c] = h[e,i]*t[e,k] (c = i*128+k),
        // or h[e,i] for the 64-wide bias extension (c >= 8192).
        v16bf a;
        const int cbase = kt * 32 + khalf * 8;
        #pragma unroll
        for (int j = 0; j < 16; ++j) {
            const int c = cbase + j + (j & 8);
            float v;
            if (c < D * EH)
                v = h_s[el * D + (c >> 7)] * t_s[el * EH + (c & 127)];
            else
                v = h_s[el * D + (c - D * EH)];
            a[j] = (__bf16)v;
        }
        if (kt + 1 < KTILES)   // L2-resident B: keep next K-tile warm
            __builtin_prefetch(&Bfrag[(size_t)(kt + 1) * NTILES * 32 + lane], 0, 1);
        #pragma unroll
        for (int nt = 0; nt < NTILES; ++nt) {
            const v16bf b = Bfrag[(size_t)(kt * NTILES + nt) * 32 + lane];
            acc[nt] = __builtin_amdgcn_wmma_f32_16x16x32_bf16(
                false, a, false, b, (short)0, acc[nt], false, false);
        }
    }

    // Segment-sum scatter: neigh[dst[e], o] += m[e, o]  (relaxed agent atomics)
    #pragma unroll
    for (int nt = 0; nt < NTILES; ++nt) {
        #pragma unroll
        for (int j = 0; j < 8; ++j) {
            const int m = j + khalf * 8;                 // C/D layout (ISA 7.12.2)
            const int e = e0 + wid * 16 + m;
            const int o = nt * 16 + (lane & 15);
            __hip_atomic_fetch_add(&neigh[(size_t)dst[e] * D + o], acc[nt][j],
                                   __ATOMIC_RELAXED, __HIP_MEMORY_SCOPE_AGENT);
        }
    }
}

// ---------------- GRU update (in-place h), also re-zeros neigh ----------------
__global__ void __launch_bounds__(64)
gru_kernel(float* __restrict__ h, float* __restrict__ neigh,
           const float* __restrict__ conv_b,
           const float* __restrict__ Wih, const float* __restrict__ Whh,
           const float* __restrict__ bih, const float* __restrict__ bhh)
{
    __shared__ float xs[D], hs[D];
    const int v = blockIdx.x, j = threadIdx.x;
    float x = neigh[(size_t)v * D + j] + conv_b[j];
    xs[j] = x > 0.f ? x : 0.f;
    hs[j] = h[(size_t)v * D + j];
    neigh[(size_t)v * D + j] = 0.f;           // ready for next step's scatter
    __syncthreads();

    float ir = bih[j], iz = bih[D + j], in_ = bih[2 * D + j];
    float hr = bhh[j], hz = bhh[D + j], hn  = bhh[2 * D + j];
    #pragma unroll 4
    for (int i = 0; i < D; ++i) {
        const float xi = xs[i], hi = hs[i];
        ir  += xi * Wih[j * D + i];
        iz  += xi * Wih[(D + j) * D + i];
        in_ += xi * Wih[(2 * D + j) * D + i];
        hr  += hi * Whh[j * D + i];
        hz  += hi * Whh[(D + j) * D + i];
        hn  += hi * Whh[(2 * D + j) * D + i];
    }
    const float r = 1.f / (1.f + __expf(-(ir + hr)));
    const float z = 1.f / (1.f + __expf(-(iz + hz)));
    const float n = tanhf(in_ + r * hn);
    h[(size_t)v * D + j] = (1.f - z) * n + z * hs[j];
}

// ---------------- predictor: score = [h[src], h[dst]] @ predW^T + b ----------------
__global__ void __launch_bounds__(256)
predict_kernel(const float* __restrict__ h, const int* __restrict__ src,
               const int* __restrict__ dst, const float* __restrict__ W,
               const float* __restrict__ b, float* __restrict__ out)
{
    const int e = blockIdx.x * 256 + threadIdx.x;
    if (e >= E_N) return;
    const float* hs = h + (size_t)src[e] * D;
    const float* hd = h + (size_t)dst[e] * D;
    float s = b[0];
    #pragma unroll 4
    for (int i = 0; i < D; ++i) s += W[i] * hs[i] + W[D + i] * hd[i];
    out[e] = s;
}

extern "C" void kernel_launch(void* const* d_in, const int* in_sizes, int n_in,
                              void* d_out, int out_size, void* d_ws, size_t ws_size,
                              hipStream_t stream)
{
    const float* node_feats = (const float*)d_in[0];
    const float* edge_feats = (const float*)d_in[1];
    const int*   src        = (const int*)  d_in[2];
    const int*   dst        = (const int*)  d_in[3];
    const float* proj_W     = (const float*)d_in[4];
    const float* proj_b     = (const float*)d_in[5];
    const float* en_W1      = (const float*)d_in[6];
    const float* en_b1      = (const float*)d_in[7];
    const float* en_W2      = (const float*)d_in[8];
    const float* en_b2      = (const float*)d_in[9];
    const float* conv_b     = (const float*)d_in[10];
    const float* gru_Wih    = (const float*)d_in[11];
    const float* gru_Whh    = (const float*)d_in[12];
    const float* gru_bih    = (const float*)d_in[13];
    const float* gru_bhh    = (const float*)d_in[14];
    const float* pred_W     = (const float*)d_in[15];
    const float* pred_b     = (const float*)d_in[16];

    // workspace: h (5.1MB) | neigh (5.1MB) | t (20.5MB) | Bpack bf16 (1.0MB)
    float*  ws    = (float*)d_ws;
    float*  h     = ws;
    float*  neigh = h + (size_t)V_N * D;
    float*  t     = neigh + (size_t)V_N * D;
    __bf16* Bpack = (__bf16*)(t + (size_t)E_N * EH);

    proj_kernel       <<<(V_N * D) / 256, 256, 0, stream>>>(node_feats, proj_W, proj_b, h);
    edge_hidden_kernel<<<(E_N * EH) / 256, 256, 0, stream>>>(edge_feats, en_W1, en_b1, t);
    pack_b_kernel     <<<(KTILES * NTILES * 512) / 256, 256, 0, stream>>>(en_W2, en_b2, Bpack);
    zero_kernel       <<<(V_N * D) / 256, 256, 0, stream>>>(neigh, V_N * D);

    for (int s = 0; s < STEPS; ++s) {
        edge_msg_kernel<<<E_N / 64, 128, 0, stream>>>(h, t, Bpack, src, dst, neigh);
        gru_kernel     <<<V_N, 64, 0, stream>>>(h, neigh, conv_b,
                                                gru_Wih, gru_Whh, gru_bih, gru_bhh);
    }
    predict_kernel<<<(E_N + 255) / 256, 256, 0, stream>>>(h, src, dst, pred_W, pred_b,
                                                          (float*)d_out);
}
